// get_model_49357764166195
// MI455X (gfx1250) — matrix operand order repaired and checked by reading, not compiled, not run on hardware
//
#include <hip/hip_runtime.h>
#include <hip/hip_bf16.h>

// DGCNN forward for MI455X (gfx1250). Edge-conv factorized into dense WMMA
// GEMMs (P = Wd*x, Q = (Wc-Wd)*x) + L2-resident gathers. Matrix math via
// v_wmma_f32_16x16x32_f16 (wave32) with pre-padded f16 operands so the GEMM
// inner loop is pure b128 loads + WMMA (no guards, no converts).

#define B_    8
#define N_    2048
#define K_    20
#define EMB_  1024
#define NCLS_ 40
#define EPS_  1e-5f
#define SLOPE_ 0.2f
#define CAT_  512                   // concat channels of x1..x4
#define NPARTS_ (B_ * (N_ / 128))   // 128 partial-stat blocks

typedef __attribute__((ext_vector_type(16))) _Float16 v16h;
typedef __attribute__((ext_vector_type(8)))  _Float16 v8h;
typedef __attribute__((ext_vector_type(8)))  float    v8f;

// ===================== kNN: per-(b,n) top-K of -||x_n - x_m||^2 =====================
__global__ void knn_kernel(const float* __restrict__ X, int C, long xbstride,
                           int* __restrict__ idx) {
  int bn = blockIdx.x;
  int b = bn / N_, n = bn % N_;
  int tid = threadIdx.x;
  __shared__ float pd[N_];
  __shared__ float xs[256];
  __shared__ float rv[256];
  __shared__ int   ri[256];
  const float* xb = X + (long)b * xbstride;
  if (tid < C) xs[tid] = xb[(long)tid * N_ + n];
  __syncthreads();
  for (int m = tid; m < N_; m += 256) {
    float acc = 0.f;
    for (int c = 0; c < C; ++c) {
      float d = xs[c] - xb[(long)c * N_ + m];
      acc = fmaf(d, d, acc);
    }
    pd[m] = -acc;                                  // pd = -dist^2 (self = 0 = max)
  }
  __syncthreads();
  for (int k = 0; k < K_; ++k) {                   // iterative argmax, ties -> lower index
    float bv = -3.0e38f; int bi = N_;
    for (int m = tid; m < N_; m += 256) {
      float v = pd[m];
      if (v > bv || (v == bv && m < bi)) { bv = v; bi = m; }
    }
    rv[tid] = bv; ri[tid] = bi;
    __syncthreads();
    for (int s = 128; s > 0; s >>= 1) {
      if (tid < s) {
        float ov = rv[tid + s]; int oi = ri[tid + s];
        if (ov > rv[tid] || (ov == rv[tid] && oi < ri[tid])) { rv[tid] = ov; ri[tid] = oi; }
      }
      __syncthreads();
    }
    if (tid == 0) {
      idx[((long)b * N_ + n) * K_ + k] = ri[0];
      pd[ri[0]] = -3.4e38f;
    }
    __syncthreads();
  }
}

// ============ stacked weight prep: rows [0,O)=Wd, [O,2O)=Wc-Wd; f16, K-padded ============
__global__ void prep_w_kernel(const float* __restrict__ w, _Float16* __restrict__ wh,
                              int O, int C, int Cpad) {
  int t = blockIdx.x * blockDim.x + threadIdx.x;
  if (t >= 2 * O * Cpad) return;
  int r = t / Cpad, c = t % Cpad;
  float v = 0.f;
  if (c < C) {
    int o = (r < O) ? r : (r - O);
    float wd = w[(long)o * 2 * C + c];
    float wc = w[(long)o * 2 * C + C + c];
    v = (r < O) ? wd : (wc - wd);
  }
  wh[(long)r * Cpad + c] = (_Float16)v;
}

// plain f32 -> f16 copy (for w5, already M2 x C with C = Cpad)
__global__ void convw_kernel(const float* __restrict__ w, _Float16* __restrict__ wh, int total) {
  int t = blockIdx.x * blockDim.x + threadIdx.x;
  if (t < total) wh[t] = (_Float16)w[t];
}

// ============ activation prep: (C, N) f32 -> (n, Cpad) f16 per batch, zero-padded ============
__global__ void prep_x_kernel(const float* __restrict__ X, long xbstride, int C, int Cpad,
                              _Float16* __restrict__ Xh) {
  int bn = blockIdx.x;
  int b = bn / N_, n = bn % N_;
  for (int c = threadIdx.x; c < Cpad; c += blockDim.x) {
    float v = (c < C) ? X[(long)b * xbstride + (long)c * N_ + n] : 0.f;
    Xh[((long)b * N_ + n) * Cpad + c] = (_Float16)v;
  }
}

// ===================== WMMA GEMM: Y[b,n,:] = Wh (M2xCpad) * Xh[b] (n,Cpad)^T =====================
// one wave per 16x16 tile; fully unrolled K-loop; pure b128 loads + v_wmma.
// ISA 7.12.2 f16 operand maps: A lane-half K-base {0,8}, elems 0-7 -> base+e, 8-15 -> base+16+e
//                              B same as A: lane-half K-base {0,8}, elems 0-7 -> base+e, 8-15 -> base+16+e
template<int CPAD>
__global__ void wmma_gemm_kernel(const _Float16* __restrict__ Wh,
                                 const _Float16* __restrict__ Xh,
                                 float* __restrict__ Y, int M2) {
  int m0 = blockIdx.x * 16;
  int n0 = blockIdx.y * 16;
  int b  = blockIdx.z;
  int lane = threadIdx.x & 31;
  int mr = lane & 15;
  const _Float16* arow = Wh + (long)(m0 + mr) * CPAD + ((lane & 16) ? 8 : 0);
  const _Float16* brow = Xh + ((long)b * N_ + n0 + mr) * CPAD + ((lane & 16) ? 8 : 0);
  v8f acc = {};
#pragma unroll
  for (int kk = 0; kk < CPAD; kk += 32) {
    v8h a0 = *(const v8h*)(arow + kk);
    v8h a1 = *(const v8h*)(arow + kk + 16);
    v16h a = __builtin_shufflevector(a0, a1, 0, 1, 2, 3, 4, 5, 6, 7,
                                             8, 9, 10, 11, 12, 13, 14, 15);
    v8h b0 = *(const v8h*)(brow + kk);
    v8h b1 = *(const v8h*)(brow + kk + 16);
    v16h bm = __builtin_shufflevector(b0, b1, 0, 1, 2, 3, 4, 5, 6, 7,
                                              8, 9, 10, 11, 12, 13, 14, 15);
    acc = __builtin_amdgcn_wmma_f32_16x16x32_f16(false, a, false, bm,
                                                 (short)0, acc, false, false);
  }
  int mBase = (lane & 16) ? 8 : 0;
  float4* yp = (float4*)&Y[((long)b * N_ + (n0 + mr)) * M2 + m0 + mBase];
  yp[0] = make_float4(acc[0], acc[1], acc[2], acc[3]);
  yp[1] = make_float4(acc[4], acc[5], acc[6], acc[7]);
}

// ===================== edge-conv BN stats (deterministic partials) =====================
// Part[part][0..1023] = sum, Part[part][1024..2047] = sumsq, part = b*(N/128)+strip
__global__ void edge_stats_kernel(const float* __restrict__ PQ, const int* __restrict__ idx,
                                  int O, float* __restrict__ Part) {
  int b = blockIdx.x, n0 = blockIdx.y * 128;
  int o = threadIdx.x;
  if (o >= O) return;
  int M2 = 2 * O;
  float s = 0.f, ss = 0.f;
  for (int n = n0; n < n0 + 128; ++n) {
    float q = PQ[((long)b * N_ + n) * M2 + O + o];
    const int* ip = idx + ((long)b * N_ + n) * K_;
    __builtin_prefetch(ip + K_, 0, 0);
    for (int k = 0; k < K_; ++k) {
      float y = PQ[((long)b * N_ + ip[k]) * M2 + o] + q;
      s += y; ss = fmaf(y, y, ss);
    }
  }
  long part = (long)b * (N_ / 128) + blockIdx.y;
  Part[part * 2048 + o] = s;
  Part[part * 2048 + 1024 + o] = ss;
}

__global__ void plain_stats_kernel(const float* __restrict__ Y, int M2,
                                   float* __restrict__ Part) {
  int b = blockIdx.x, n0 = blockIdx.y * 128;
  long part = (long)b * (N_ / 128) + blockIdx.y;
  for (int o = threadIdx.x; o < M2; o += blockDim.x) {
    float s = 0.f, ss = 0.f;
    for (int n = n0; n < n0 + 128; ++n) {
      float y = Y[((long)b * N_ + n) * M2 + o];
      s += y; ss = fmaf(y, y, ss);
    }
    Part[part * 2048 + o] = s;
    Part[part * 2048 + 1024 + o] = ss;
  }
}

__global__ void finalize_kernel(const float* __restrict__ Part, int nparts,
                                const float* __restrict__ g, const float* __restrict__ bt,
                                float* __restrict__ aArr, float* __restrict__ cArr,
                                int O, float invcnt) {
  int o = blockIdx.x * blockDim.x + threadIdx.x;
  if (o >= O) return;
  float s = 0.f, ss = 0.f;
  for (int pi = 0; pi < nparts; ++pi) {
    s  += Part[(long)pi * 2048 + o];
    ss += Part[(long)pi * 2048 + 1024 + o];
  }
  float mean = s * invcnt;
  float var  = ss * invcnt - mean * mean;
  float a = g[o] * rsqrtf(var + EPS_);
  aArr[o] = a;
  cArr[o] = bt[o] - mean * a;
}

// ===================== edge apply: bn affine + lrelu + max over k =====================
__global__ void edge_apply_kernel(const float* __restrict__ PQ, const int* __restrict__ idx,
                                  const float* __restrict__ aArr, const float* __restrict__ cArr,
                                  int O, float* __restrict__ xcat, int co) {
  int b = blockIdx.x, n0 = blockIdx.y * 128;
  int o = threadIdx.x;
  if (o >= O) return;
  int M2 = 2 * O;
  float a = aArr[o], cc = cArr[o];
  for (int n = n0; n < n0 + 128; ++n) {
    float q = PQ[((long)b * N_ + n) * M2 + O + o];
    const int* ip = idx + ((long)b * N_ + n) * K_;
    float best = -3.4e38f;
    for (int k = 0; k < K_; ++k) {
      float y = fmaf(a, PQ[((long)b * N_ + ip[k]) * M2 + o] + q, cc);
      float z = (y >= 0.f) ? y : SLOPE_ * y;
      best = fmaxf(best, z);
    }
    xcat[((long)b * CAT_ + co + o) * N_ + n] = best;
  }
}

// ===================== global max+mean pool after conv5 bn/lrelu =====================
__global__ void pool_kernel(const float* __restrict__ Y, const float* __restrict__ aArr,
                            const float* __restrict__ cArr, float* __restrict__ p) {
  int t = blockIdx.x * blockDim.x + threadIdx.x;
  if (t >= B_ * EMB_) return;
  int b = t / EMB_, o = t % EMB_;
  float a = aArr[o], cc = cArr[o];
  float mx = -3.4e38f, sm = 0.f;
  for (int n = 0; n < N_; ++n) {
    float y = fmaf(a, Y[((long)b * N_ + n) * EMB_ + o], cc);
    float z = (y >= 0.f) ? y : SLOPE_ * y;
    mx = fmaxf(mx, z); sm += z;
  }
  p[(long)b * 2 * EMB_ + o] = mx;
  p[(long)b * 2 * EMB_ + EMB_ + o] = sm * (1.f / N_);
}

// ===================== head FCs =====================
__global__ void fc_kernel(const float* __restrict__ A, const float* __restrict__ W,
                          const float* __restrict__ bias, float* __restrict__ Y,
                          int In, int Out) {
  int t = blockIdx.x * blockDim.x + threadIdx.x;
  if (t >= B_ * Out) return;
  int b = t / Out, j = t % Out;
  float s = bias ? bias[j] : 0.f;
  const float* ar = A + (long)b * In;
  const float* wr = W + (long)j * In;
  for (int i = 0; i < In; ++i) s = fmaf(ar[i], wr[i], s);
  Y[t] = s;
}

__global__ void bnrow_kernel(const float* __restrict__ Yin, const float* __restrict__ g,
                             const float* __restrict__ bt, float* __restrict__ Yout, int Out) {
  int j = blockIdx.x * blockDim.x + threadIdx.x;
  if (j >= Out) return;
  float s = 0.f, ss = 0.f;
  for (int b = 0; b < B_; ++b) { float v = Yin[b * Out + j]; s += v; ss = fmaf(v, v, ss); }
  float mean = s * (1.f / B_);
  float var  = ss * (1.f / B_) - mean * mean;
  float a = g[j] * rsqrtf(var + EPS_);
  float cc = bt[j] - mean * a;
  for (int b = 0; b < B_; ++b) {
    float y = fmaf(a, Yin[b * Out + j], cc);
    Yout[b * Out + j] = (y >= 0.f) ? y : SLOPE_ * y;
  }
}

// ===================== launch =====================
extern "C" void kernel_launch(void* const* d_in, const int* in_sizes, int n_in,
                              void* d_out, int out_size, void* d_ws, size_t ws_size,
                              hipStream_t stream) {
  const float* x   = (const float*)d_in[0];
  const float* w1  = (const float*)d_in[1];
  const float* w2  = (const float*)d_in[2];
  const float* w3  = (const float*)d_in[3];
  const float* w4  = (const float*)d_in[4];
  const float* w5  = (const float*)d_in[5];
  const float* lw1 = (const float*)d_in[6];
  const float* lw2 = (const float*)d_in[7];
  const float* lb2 = (const float*)d_in[8];
  const float* lw3 = (const float*)d_in[9];
  const float* lb3 = (const float*)d_in[10];
  const float* gArr[8], *bArr[8];
  for (int i = 1; i <= 7; ++i) {
    gArr[i] = (const float*)d_in[11 + 2 * (i - 1)];
    bArr[i] = (const float*)d_in[12 + 2 * (i - 1)];
  }

  char* ws = (char*)d_ws;
  size_t off = 0;
  auto carve = [&](size_t bytes) { void* pp = ws + off; off += (bytes + 255) & ~(size_t)255; return pp; };
  int*       idx  = (int*)      carve((size_t)B_ * N_ * K_ * 4);
  float*     xcat = (float*)    carve((size_t)B_ * CAT_ * N_ * 4);
  float*     PQ   = (float*)    carve((size_t)B_ * N_ * EMB_ * 4);
  _Float16*  Wh   = (_Float16*) carve((size_t)EMB_ * CAT_ * 2);
  _Float16*  Xh   = (_Float16*) carve((size_t)B_ * N_ * CAT_ * 2);
  float*     Part = (float*)    carve((size_t)NPARTS_ * 2048 * 4);
  float*     aArr = (float*)    carve(EMB_ * 4);
  float*     cArr = (float*)    carve(EMB_ * 4);
  float*     p    = (float*)    carve((size_t)B_ * 2 * EMB_ * 4);
  float*     t1   = (float*)    carve((size_t)B_ * 512 * 4);
  float*     t2   = (float*)    carve((size_t)B_ * 512 * 4);
  float*     t3   = (float*)    carve((size_t)B_ * 256 * 4);
  float*     t4   = (float*)    carve((size_t)B_ * 256 * 4);
  (void)ws_size; (void)in_sizes; (void)n_in; (void)out_size;

  struct Blk { const float* w; int gi; int Cin; int Cpad; int O; int co; const float* xin; long xbs; };
  Blk blks[4] = {
    { w1, 1,   3,  32,  64,   0, x,               (long)3 * N_ },
    { w2, 2,  64,  64,  64,  64, xcat,            (long)CAT_ * N_ },
    { w3, 3,  64,  64, 128, 128, xcat +  64 * N_, (long)CAT_ * N_ },
    { w4, 4, 128, 128, 256, 256, xcat + 128 * N_, (long)CAT_ * N_ },
  };

  for (int bi = 0; bi < 4; ++bi) {
    const Blk& bk = blks[bi];
    knn_kernel<<<B_ * N_, 256, 0, stream>>>(bk.xin, bk.Cin, bk.xbs, idx);
    int wcnt = 2 * bk.O * bk.Cpad;
    prep_w_kernel<<<(wcnt + 255) / 256, 256, 0, stream>>>(bk.w, Wh, bk.O, bk.Cin, bk.Cpad);
    prep_x_kernel<<<B_ * N_, 128, 0, stream>>>(bk.xin, bk.xbs, bk.Cin, bk.Cpad, Xh);
    dim3 gg(2 * bk.O / 16, N_ / 16, B_);
    if (bk.Cpad == 32)
      wmma_gemm_kernel<32><<<gg, 32, 0, stream>>>(Wh, Xh, PQ, 2 * bk.O);
    else if (bk.Cpad == 64)
      wmma_gemm_kernel<64><<<gg, 32, 0, stream>>>(Wh, Xh, PQ, 2 * bk.O);
    else
      wmma_gemm_kernel<128><<<gg, 32, 0, stream>>>(Wh, Xh, PQ, 2 * bk.O);
    edge_stats_kernel<<<dim3(B_, N_ / 128), 256, 0, stream>>>(PQ, idx, bk.O, Part);
    finalize_kernel<<<(bk.O + 255) / 256, 256, 0, stream>>>(
        Part, NPARTS_, gArr[bk.gi], bArr[bk.gi], aArr, cArr, bk.O,
        1.f / ((float)B_ * N_ * K_));
    edge_apply_kernel<<<dim3(B_, N_ / 128), 256, 0, stream>>>(PQ, idx, aArr, cArr, bk.O, xcat, bk.co);
  }

  // conv5: 1024 x 512 over xcat
  convw_kernel<<<(EMB_ * CAT_ + 255) / 256, 256, 0, stream>>>(w5, Wh, EMB_ * CAT_);
  prep_x_kernel<<<B_ * N_, 128, 0, stream>>>(xcat, (long)CAT_ * N_, CAT_, CAT_, Xh);
  dim3 gg5(EMB_ / 16, N_ / 16, B_);
  wmma_gemm_kernel<512><<<gg5, 32, 0, stream>>>(Wh, Xh, PQ, EMB_);
  plain_stats_kernel<<<dim3(B_, N_ / 128), 256, 0, stream>>>(PQ, EMB_, Part);
  finalize_kernel<<<(EMB_ + 255) / 256, 256, 0, stream>>>(
      Part, NPARTS_, gArr[5], bArr[5], aArr, cArr, EMB_, 1.f / ((float)B_ * N_));
  pool_kernel<<<(B_ * EMB_ + 255) / 256, 256, 0, stream>>>(PQ, aArr, cArr, p);

  // head MLP
  fc_kernel<<<(B_ * 512 + 255) / 256, 256, 0, stream>>>(p, lw1, nullptr, t1, 2 * EMB_, 512);
  bnrow_kernel<<<2, 256, 0, stream>>>(t1, gArr[6], bArr[6], t2, 512);
  fc_kernel<<<(B_ * 256 + 255) / 256, 256, 0, stream>>>(t2, lw2, lb2, t3, 512, 256);
  bnrow_kernel<<<1, 256, 0, stream>>>(t3, gArr[7], bArr[7], t4, 256);
  fc_kernel<<<(B_ * NCLS_ + 255) / 256, 256, 0, stream>>>(t4, lw3, lb3, (float*)d_out, 256, NCLS_);
}
